// GATGCN_40175124087240
// MI455X (gfx1250) — matrix operand order repaired and checked by reading, not compiled, not run on hardware
//
#include <hip/hip_runtime.h>
#include <hip/hip_bf16.h>

// ---------------------------------------------------------------------------
// GATv2 (single head, D=128) + 3-layer MLP for MI455X (gfx1250, wave32, WMMA)
// GEMMs: LDS-staged A tiles + packed-B fragments + v_wmma_f32_16x16x32_bf16.
// ---------------------------------------------------------------------------

typedef __bf16 bf16_t;
typedef __attribute__((ext_vector_type(16))) __bf16 v16bf;
typedef __attribute__((ext_vector_type(8)))  __bf16 bf16x8;
typedef __attribute__((ext_vector_type(8)))  float  v8f;

#define DIN_       128
#define NEG_SLOPE  0.2f
#define LDS_STRIDE 136   // 128 + 8 bf16 pad: row r -> bank offset 4r (conflict-free b128)

// ---------------------------------------------------------------------------
// Pack a row-major f32 weight matrix W[K][Nout] into the per-lane WMMA
// B-fragment layout for v_wmma_f32_16x16x32_bf16:
//   fragment index f = kc * ntiles + nb   (kc: K/32 chunk, nb: N/16 tile)
//   Bp[(f*32 + lane)*16 + i] = W[kc*32 + (i&7) + (i>>3)*16 + (lane>>4)*8][nb*16 + (lane&15)]
// ---------------------------------------------------------------------------
__global__ void pack_b_kernel(const float* __restrict__ W,
                              bf16_t* __restrict__ Bp,
                              int K, int Nout)
{
    int t = blockIdx.x * blockDim.x + threadIdx.x;
    int total = K * Nout;
    if (t >= total) return;
    int i    = t & 15;
    int lane = (t >> 4) & 31;
    int f    = t >> 9;
    int ntiles = Nout >> 4;
    int nb = f % ntiles;
    int kc = f / ntiles;
    int k  = (kc << 5) + (i & 7) + ((i >> 3) << 4) + ((lane >> 4) << 3);
    int n  = (nb << 4) + (lane & 15);
    Bp[t] = (bf16_t)W[k * Nout + n];
}

// f32 -> bf16 elementwise
__global__ void cvt_f32_bf16_kernel(const float* __restrict__ src,
                                    bf16_t* __restrict__ dst, int n)
{
    int t = blockIdx.x * blockDim.x + threadIdx.x;
    if (t < n) dst[t] = (bf16_t)src[t];
}

// ---------------------------------------------------------------------------
// WMMA GEMM:  C[M x Nout] = A[M x K](bf16) * W(packed bf16) + bias, optional
// ReLU, optional f32 and/or bf16 outputs.
//
// One 128-thread block (4 waves) per 16-row m-tile. The block stages the
// 16 x K A-tile in LDS once (coalesced b128 loads), then each wave computes
// NTPW = NTILES/4 n-tiles. Fully unrolled over K/32 chunks so all loads use
// immediate offsets. Compile-time: K % 32 == 0, NTILES in {4, 8}.
// ---------------------------------------------------------------------------
template <int NTILES, int K>
__global__ __launch_bounds__(128)
void wmma_gemm_kernel(const bf16_t* __restrict__ A,
                      const bf16_t* __restrict__ Bp,
                      const float*  __restrict__ bias,
                      float*  __restrict__ Cf,
                      bf16_t* __restrict__ Cb,
                      int M, int relu)
{
    constexpr int NOUT = NTILES * 16;
    constexpr int NTPW = (NTILES >= 8) ? 2 : 1;  // n-tiles per wave
    constexpr int KCN  = K >> 5;                 // K=32 chunks

    __shared__ __align__(16) bf16_t Atile[16 * LDS_STRIDE];

    const int tid  = threadIdx.x;        // 0..127
    const int lane = tid & 31;
    const int wv   = tid >> 5;           // wave id 0..3
    const int m0   = blockIdx.x << 4;

    // ---- cooperative, coalesced stage of the 16 x K A-tile into LDS ----
    {
        const int row   = tid >> 3;      // 0..15
        const int chunk = tid & 7;       // 8-elem (16B) chunks along K
        int am = m0 + row;
        if (am >= M) am = M - 1;         // clamp: duplicate row, harmless
        const bf16_t* src = A + (size_t)am * K;
#pragma unroll
        for (int c = chunk; c < (K >> 3); c += 8) {
            *(bf16x8*)(&Atile[row * LDS_STRIDE + (c << 3)]) =
                *(const bf16x8*)(src + (c << 3));
        }
    }
    __syncthreads();

    const int nt0 = wv * NTPW;
    const int hl  = lane >> 4;
    const int col = lane & 15;

    v8f acc0, acc1;
    {
        const float b0 = bias[(nt0 << 4) + col];
        const float b1 = (NTPW == 2) ? bias[((nt0 + 1) << 4) + col] : 0.f;
#pragma unroll
        for (int r = 0; r < 8; ++r) { acc0[r] = b0; acc1[r] = b1; }
    }

    // Base pointers: per-kc stride is NTILES*512 elements -> immediate offsets
    const bf16_t* arow = &Atile[col * LDS_STRIDE];
    const bf16_t* bp0  = Bp + ((size_t)nt0 * 32 + lane) * 16;
    const bf16_t* bp1  = Bp + ((size_t)(nt0 + 1) * 32 + lane) * 16;

    // A-fragment: lane L holds row (L&15); within a K=32 chunk,
    // element i -> k = (i<8 ? i : i+8) + hl*8
#pragma unroll
    for (int kc = 0; kc < KCN; ++kc) {
        const int kb = kc << 5;
        bf16x8 alo = *(const bf16x8*)(arow + kb + hl * 8);
        bf16x8 ahi = *(const bf16x8*)(arow + kb + 16 + hl * 8);
        v16bf av = __builtin_shufflevector(alo, ahi,
                    0,1,2,3,4,5,6,7,8,9,10,11,12,13,14,15);
        v16bf bv0 = *(const v16bf*)(bp0 + (size_t)kc * NTILES * 512);
        acc0 = __builtin_amdgcn_wmma_f32_16x16x32_bf16(
                   false, av, false, bv0, (short)0, acc0, false, false);
        if (NTPW == 2) {
            v16bf bv1 = *(const v16bf*)(bp1 + (size_t)kc * NTILES * 512);
            acc1 = __builtin_amdgcn_wmma_f32_16x16x32_bf16(
                       false, av, false, bv1, (short)0, acc1, false, false);
        }
    }

    // C/D layout: lane L, vgpr r -> row m0 + r + hl*8, col = L&15
#pragma unroll
    for (int r = 0; r < 8; ++r) {
        int row = m0 + r + hl * 8;
        if (row < M) {
            float v0 = acc0[r];
            if (relu) v0 = v0 > 0.f ? v0 : 0.f;
            size_t idx0 = (size_t)row * NOUT + (nt0 << 4) + col;
            if (Cf) Cf[idx0] = v0;
            if (Cb) Cb[idx0] = (bf16_t)v0;
            if (NTPW == 2) {
                float v1 = acc1[r];
                if (relu) v1 = v1 > 0.f ? v1 : 0.f;
                size_t idx1 = (size_t)row * NOUT + ((nt0 + 1) << 4) + col;
                if (Cf) Cf[idx1] = v1;
                if (Cb) Cb[idx1] = (bf16_t)v1;
            }
        }
    }
}

// ---------------------------------------------------------------------------
// init: h[n][d] = bg[d]; amax[n] = -FLT_MAX; denom[n] = 0
// ---------------------------------------------------------------------------
__global__ void init_kernel(float* __restrict__ h, const float* __restrict__ bg,
                            float* __restrict__ amax, float* __restrict__ denom,
                            int Nn)
{
    int t = blockIdx.x * blockDim.x + threadIdx.x;
    if (t >= Nn * DIN_) return;
    int d = t & (DIN_ - 1);
    h[t] = bg[d];
    if (d == 0) {
        int n = t >> 7;
        amax[n]  = -3.402823466e38f;
        denom[n] = 0.f;
    }
}

__device__ __forceinline__ void atomic_max_float(float* addr, float val)
{
    if (val >= 0.f) atomicMax((int*)addr, __float_as_int(val));
    else            atomicMin((unsigned int*)addr, __float_as_uint(val));
}

// ---------------------------------------------------------------------------
// Edge scores: a[e] = leaky_relu(xl[src] + xr[dst]) . att ; atomic max per dst
// One wave32 per edge; lane handles 4 of 128 dims.
// ---------------------------------------------------------------------------
__global__ __launch_bounds__(256)
void edge_score_kernel(const long long* __restrict__ ei,
                       const float* __restrict__ xl,
                       const float* __restrict__ xr,
                       const float* __restrict__ att,
                       float* __restrict__ a,
                       float* __restrict__ amax,
                       int E, int Nn)
{
    int gw   = (blockIdx.x * blockDim.x + threadIdx.x) >> 5;  // uniform per wave
    int lane = threadIdx.x & 31;
    int EN = E + Nn;
    if (gw >= EN) return;

    long long j, i;
    if (gw < E) { j = ei[gw]; i = ei[E + gw]; }
    else        { j = i = (long long)(gw - E); }

    float4 lv = ((const float4*)(xl + (size_t)j * DIN_))[lane];
    float4 rv = ((const float4*)(xr + (size_t)i * DIN_))[lane];
    float4 av = ((const float4*)att)[lane];

    float t0 = lv.x + rv.x; t0 = t0 > 0.f ? t0 : NEG_SLOPE * t0;
    float t1 = lv.y + rv.y; t1 = t1 > 0.f ? t1 : NEG_SLOPE * t1;
    float t2 = lv.z + rv.z; t2 = t2 > 0.f ? t2 : NEG_SLOPE * t2;
    float t3 = lv.w + rv.w; t3 = t3 > 0.f ? t3 : NEG_SLOPE * t3;
    float s = t0 * av.x + t1 * av.y + t2 * av.z + t3 * av.w;

#pragma unroll
    for (int off = 16; off >= 1; off >>= 1) s += __shfl_xor(s, off, 32);

    if (lane == 0) {
        a[gw] = s;
        atomic_max_float(amax + i, s);
    }
}

// ---------------------------------------------------------------------------
// ea[e] = exp(a[e] - amax[dst]); denom[dst] += ea  (one thread per edge)
// ---------------------------------------------------------------------------
__global__ void edge_exp_kernel(const long long* __restrict__ ei,
                                float* __restrict__ a,
                                const float* __restrict__ amax,
                                float* __restrict__ denom,
                                int E, int Nn)
{
    int e = blockIdx.x * blockDim.x + threadIdx.x;
    int EN = E + Nn;
    if (e >= EN) return;
    long long i = (e < E) ? ei[E + e] : (long long)(e - E);
    float v = __expf(a[e] - amax[i]);
    a[e] = v;
    atomicAdd(denom + i, v);
}

// ---------------------------------------------------------------------------
// h[dst] += (ea[e]/denom[dst]) * xl[src]  (wave per edge, 4 dims/lane)
// ---------------------------------------------------------------------------
__global__ __launch_bounds__(256)
void aggregate_kernel(const long long* __restrict__ ei,
                      const float* __restrict__ a,
                      const float* __restrict__ denom,
                      const float* __restrict__ xl,
                      float* __restrict__ h,
                      int E, int Nn)
{
    int gw   = (blockIdx.x * blockDim.x + threadIdx.x) >> 5;
    int lane = threadIdx.x & 31;
    int EN = E + Nn;
    if (gw >= EN) return;

    long long j, i;
    if (gw < E) { j = ei[gw]; i = ei[E + gw]; }
    else        { j = i = (long long)(gw - E); }

    float alpha = a[gw] / denom[i];
    float4 lv = ((const float4*)(xl + (size_t)j * DIN_))[lane];
    float* hp = h + (size_t)i * DIN_ + lane * 4;
    atomicAdd(hp + 0, alpha * lv.x);
    atomicAdd(hp + 1, alpha * lv.y);
    atomicAdd(hp + 2, alpha * lv.z);
    atomicAdd(hp + 3, alpha * lv.w);
}

// ---------------------------------------------------------------------------
// Host-side launch
// ---------------------------------------------------------------------------
static inline char* carve(char*& p, size_t bytes)
{
    char* r = p;
    p += (bytes + 255) & ~(size_t)255;
    return r;
}

extern "C" void kernel_launch(void* const* d_in, const int* in_sizes, int n_in,
                              void* d_out, int out_size, void* d_ws, size_t ws_size,
                              hipStream_t stream)
{
    const float*     x   = (const float*)d_in[0];
    const long long* ei  = (const long long*)d_in[1];
    const float*     Wl  = (const float*)d_in[2];
    const float*     bl  = (const float*)d_in[3];
    const float*     Wr  = (const float*)d_in[4];
    const float*     br  = (const float*)d_in[5];
    const float*     att = (const float*)d_in[6];
    const float*     bg  = (const float*)d_in[7];
    const float*     W1  = (const float*)d_in[8];
    const float*     b1  = (const float*)d_in[9];
    const float*     W2  = (const float*)d_in[10];
    const float*     b2  = (const float*)d_in[11];
    const float*     W3  = (const float*)d_in[12];
    const float*     b3  = (const float*)d_in[13];
    float* out = (float*)d_out;

    const int Nn = in_sizes[0] / DIN_;     // 50000
    const int E  = in_sizes[1] / 2;        // 800000
    const int EN = E + Nn;
    const int HIDc = 128, DOUTc = 64;

    // ---- workspace carve ----
    char* p = (char*)d_ws;
    bf16_t* xb   = (bf16_t*)carve(p, (size_t)Nn * DIN_ * 2);
    float*  xl   = (float*) carve(p, (size_t)Nn * DIN_ * 4);
    float*  xr   = (float*) carve(p, (size_t)Nn * DIN_ * 4);
    float*  a    = (float*) carve(p, (size_t)EN * 4);
    float*  amax = (float*) carve(p, (size_t)Nn * 4);
    float*  den  = (float*) carve(p, (size_t)Nn * 4);
    float*  h    = (float*) carve(p, (size_t)Nn * DIN_ * 4);
    bf16_t* hb   = (bf16_t*)carve(p, (size_t)Nn * DIN_ * 2);
    bf16_t* h1b  = (bf16_t*)carve(p, (size_t)Nn * HIDc * 2);
    bf16_t* h2b  = (bf16_t*)carve(p, (size_t)Nn * HIDc * 2);
    bf16_t* Wlp  = (bf16_t*)carve(p, (size_t)DIN_ * DIN_ * 2);
    bf16_t* Wrp  = (bf16_t*)carve(p, (size_t)DIN_ * DIN_ * 2);
    bf16_t* W1p  = (bf16_t*)carve(p, (size_t)DIN_ * HIDc * 2);
    bf16_t* W2p  = (bf16_t*)carve(p, (size_t)HIDc * HIDc * 2);
    bf16_t* W3p  = (bf16_t*)carve(p, (size_t)HIDc * DOUTc * 2);

    // ---- 1) pack weights to bf16 fragment layout ----
    pack_b_kernel<<<(DIN_ * DIN_ + 255) / 256, 256, 0, stream>>>(Wl, Wlp, DIN_, DIN_);
    pack_b_kernel<<<(DIN_ * DIN_ + 255) / 256, 256, 0, stream>>>(Wr, Wrp, DIN_, DIN_);
    pack_b_kernel<<<(DIN_ * HIDc + 255) / 256, 256, 0, stream>>>(W1, W1p, DIN_, HIDc);
    pack_b_kernel<<<(HIDc * HIDc + 255) / 256, 256, 0, stream>>>(W2, W2p, HIDc, HIDc);
    pack_b_kernel<<<(HIDc * DOUTc + 255) / 256, 256, 0, stream>>>(W3, W3p, HIDc, DOUTc);

    // ---- 2) x -> bf16 ----
    cvt_f32_bf16_kernel<<<((size_t)Nn * DIN_ + 255) / 256, 256, 0, stream>>>(x, xb, Nn * DIN_);

    // ---- 3) xl = x@Wl + bl ; xr = x@Wr + br  (WMMA, LDS-staged A) ----
    int mtiles = (Nn + 15) / 16;
    wmma_gemm_kernel<8, 128><<<mtiles, 128, 0, stream>>>(xb, Wlp, bl, xl, nullptr, Nn, 0);
    wmma_gemm_kernel<8, 128><<<mtiles, 128, 0, stream>>>(xb, Wrp, br, xr, nullptr, Nn, 0);

    // ---- 4) init h = bg (broadcast), amax = -inf, denom = 0 ----
    init_kernel<<<((size_t)Nn * DIN_ + 255) / 256, 256, 0, stream>>>(h, bg, amax, den, Nn);

    // ---- 5) edge scores + segment max ----
    size_t escore_threads = (size_t)EN * 32;
    edge_score_kernel<<<(escore_threads + 255) / 256, 256, 0, stream>>>(ei, xl, xr, att, a, amax, E, Nn);

    // ---- 6) exp + segment sum ----
    edge_exp_kernel<<<(EN + 255) / 256, 256, 0, stream>>>(ei, a, amax, den, E, Nn);

    // ---- 7) weighted aggregation into h ----
    aggregate_kernel<<<(escore_threads + 255) / 256, 256, 0, stream>>>(ei, a, den, xl, h, E, Nn);

    // ---- 8) h -> bf16 ----
    cvt_f32_bf16_kernel<<<((size_t)Nn * DIN_ + 255) / 256, 256, 0, stream>>>(h, hb, Nn * DIN_);

    // ---- 9) MLP: relu(h@W1+b1), relu(@W2+b2), @W3+b3 -> out ----
    wmma_gemm_kernel<8, 128><<<mtiles, 128, 0, stream>>>(hb,  W1p, b1, nullptr, h1b, Nn, 1);
    wmma_gemm_kernel<8, 128><<<mtiles, 128, 0, stream>>>(h1b, W2p, b2, nullptr, h2b, Nn, 1);
    wmma_gemm_kernel<4, 128><<<mtiles, 128, 0, stream>>>(h2b, W3p, b3, out, nullptr, Nn, 0);
}
